// QuantLinearFP8_62861141344487
// MI455X (gfx1250) — matrix operand order repaired and checked by reading, not compiled, not run on hardware
//
#include <hip/hip_runtime.h>

typedef __attribute__((ext_vector_type(16))) _Float16 v16h;
typedef __attribute__((ext_vector_type(8)))  _Float16 v8h;
typedef __attribute__((ext_vector_type(4)))  _Float16 v4h;
typedef __attribute__((ext_vector_type(8)))  float    v8f;
typedef __attribute__((ext_vector_type(4)))  float    v4f;

#define M_DIM 64
#define K_DIM 4096
#define N_DIM 11008
#define GROUP 128
#define NGRP  32   // K_DIM / GROUP

// ---------------------------------------------------------------------------
// Pass 1: convert x (f32, 64x4096 row-major) -> f16 in workspace, same layout.
// 1 MB read / 0.5 MB written once; negligible vs the 180 MB weight stream.
// ---------------------------------------------------------------------------
__global__ __launch_bounds__(256) void xcvt_kernel(const float* __restrict__ x,
                                                   _Float16* __restrict__ xh,
                                                   int n) {
  int i = (blockIdx.x * 256 + threadIdx.x) * 4;
  if (i + 3 < n) {
    v4f v = *(const v4f*)(x + i);
    v4h h;
    h[0] = (_Float16)v[0];
    h[1] = (_Float16)v[1];
    h[2] = (_Float16)v[2];
    h[3] = (_Float16)v[3];
    *(v4h*)(xh + i) = h;
  }
}

// ---------------------------------------------------------------------------
// Pass 2: fused dequant GEMM.
//   grid.x = N/16 = 688 n-stripes, block = 128 threads = 4 wave32s.
//   wave w computes the 16x16 output tile (m-tile = w, n-stripe = blockIdx.x)
//   looping over all K. Scales applied once per 128-k group on the f32
//   accumulator (distributive over the group partial sum).
//
// WMMA f16 16x16x32 fragment layouts (wave32):
//   A (16x32): lane(0-15) row m; VGPR0-3 hold K=half*8..+7, VGPR4-7 K=16+half*8..+7
//   B (32x16): lane holds column n = lane%16; this code puts K = half*16 + 2i,2i+1 in VGPR i, for A as well
//   C (16x16): VGPR j -> M = half*8 + j, N = lane%16
// ---------------------------------------------------------------------------
__global__ __launch_bounds__(128) void qgemm_kernel(const _Float16* __restrict__ xh,
                                                    const float* __restrict__ w,
                                                    const float* __restrict__ wscale,
                                                    const float* __restrict__ bias,
                                                    float* __restrict__ out) {
  const int lane = threadIdx.x & 31;
  const int wave = threadIdx.x >> 5;
  const int col  = lane & 15;   // n within stripe / m within tile
  const int half = lane >> 4;   // k-half selector

  const int n0 = blockIdx.x * 16;
  const int n  = n0 + col;      // this lane's output column (B fragment column)
  const int m0 = wave * 16;     // this wave's m-tile

  // A: lane holds row (m0+col); two contiguous 8-half runs per k-step.
  const _Float16* aRow = xh + (size_t)(m0 + col) * K_DIM;
  // B: lane streams contiguous k of weight row n.
  const float* bRow = w + (size_t)n * K_DIM;
  const float* sRow = wscale + (size_t)n * NGRP;

  v8f acc = {};

  for (int g = 0; g < NGRP; ++g) {
    if (g + 1 < NGRP) {
      // Pull next group's weight bytes toward the caches (NT stream).
      __builtin_prefetch(bRow + (g + 1) * GROUP + half * 64, 0, 1);
    }

    v8f gacc = {};
#pragma unroll
    for (int t = 0; t < 4; ++t) {
      const int k0 = g * GROUP + t * 32;

      // ---- A fragment: f16 direct loads (16B each) ----
      v8h a_lo = *(const v8h*)(aRow + k0 + half * 16);
      v8h a_hi = *(const v8h*)(aRow + k0 + half * 16 + 8);
      v16h a;
#pragma unroll
      for (int i = 0; i < 8; ++i) { a[i] = a_lo[i]; a[8 + i] = a_hi[i]; }

      // ---- B fragment: 16 contiguous f32 (non-temporal) -> f16 ----
      const v4f* bp = (const v4f*)(bRow + k0 + half * 16);
      v16h b;
#pragma unroll
      for (int i = 0; i < 4; ++i) {
        v4f bv = __builtin_nontemporal_load(bp + i);
        b[4 * i + 0] = (_Float16)bv[0];
        b[4 * i + 1] = (_Float16)bv[1];
        b[4 * i + 2] = (_Float16)bv[2];
        b[4 * i + 3] = (_Float16)bv[3];
      }

      gacc = __builtin_amdgcn_wmma_f32_16x16x32_f16(
          /*neg_a=*/false, a, /*neg_b=*/false, b,
          /*c_mod=*/(short)0, gacc, /*reuse_a=*/false, /*reuse_b=*/false);
    }

    // Dequant: scale the group partial into the running accumulator.
    const float s = sRow[g];
#pragma unroll
    for (int j = 0; j < 8; ++j) acc[j] += gacc[j] * s;
  }

  // Epilogue: bias + store per C layout (row = m0 + half*8 + j, col = n).
  const float bb = bias[n];
#pragma unroll
  for (int j = 0; j < 8; ++j) {
    const int row = m0 + half * 8 + j;
    out[(size_t)row * N_DIM + n] = acc[j] + bb;
  }
}

extern "C" void kernel_launch(void* const* d_in, const int* in_sizes, int n_in,
                              void* d_out, int out_size, void* d_ws, size_t ws_size,
                              hipStream_t stream) {
  const float* x      = (const float*)d_in[0];   // (64, 4096) f32
  const float* weight = (const float*)d_in[1];   // (11008, 4096) f32 (fp8-representable)
  const float* wscale = (const float*)d_in[2];   // (11008, 32) f32
  const float* bias   = (const float*)d_in[3];   // (11008,) f32
  float* out = (float*)d_out;                    // (64, 11008) f32

  _Float16* xh = (_Float16*)d_ws;                // 64*4096*2 = 512 KB scratch

  const int nx = M_DIM * K_DIM;                  // 262144
  xcvt_kernel<<<(nx / 4 + 255) / 256, 256, 0, stream>>>(x, xh, nx);
  qgemm_kernel<<<N_DIM / 16, 128, 0, stream>>>(xh, weight, wscale, bias, out);
}
